// CrossAttention_65893388255522
// MI455X (gfx1250) — matrix operand order repaired and checked by reading, not compiled, not run on hardware
//
#include <hip/hip_runtime.h>

// ---------------------------------------------------------------------------
// CrossAttention for MI455X (gfx1250): bf16 WMMA GEMMs (async-LDS, software
// pipelined, constant-buffer double buffering) + fused flash-attention.
//   B=2, N=2048, C=1024, H=16, D=64
// ---------------------------------------------------------------------------

typedef unsigned short ushort_t;
typedef __bf16 v16bf __attribute__((ext_vector_type(16)));
typedef __bf16 v8bf  __attribute__((ext_vector_type(8)));
typedef float  v8f   __attribute__((ext_vector_type(8)));

#define SHUF16(a, b) __builtin_shufflevector((a), (b), 0,1,2,3,4,5,6,7,8,9,10,11,12,13,14,15)
#define WMMA_BF16(a, b, c) \
  __builtin_amdgcn_wmma_f32_16x16x32_bf16(false, (a), false, (b), (short)0, (c), false, false)

#define BB 2
#define NN 2048
#define CC 1024
#define HH 16
#define DD 64

static __device__ __forceinline__ ushort_t f2bf(float f) {
  unsigned int u = __float_as_uint(f);
  unsigned int r = ((u >> 16) & 1u) + 0x7FFFu;   // round-to-nearest-even
  return (ushort_t)((u + r) >> 16);
}

// LDS byte address of a generic pointer to __shared__ (low 32 bits of the
// shared-aperture generic address are the LDS byte offset).
static __device__ __forceinline__ unsigned lds_off(const void* p) {
  return (unsigned)(unsigned long long)p;
}

// One per-lane 16B global -> LDS async copy (ASYNCcnt-tracked DMA path).
static __device__ __forceinline__ void async_cp16(const void* g, unsigned lds_byte) {
  asm volatile("global_load_async_to_lds_b128 %0, %1, off"
               :: "v"(lds_byte), "v"(g)
               : "memory");
}

// ---------------------------------------------------------------------------
// fp32 -> bf16 conversion (done once for activations + weights)
// ---------------------------------------------------------------------------
__global__ __launch_bounds__(256) void cvt_f32_bf16(const float* __restrict__ in,
                                                    ushort_t* __restrict__ out, int n) {
  int i = blockIdx.x * 256 + threadIdx.x;
  if (i < n) out[i] = f2bf(in[i]);
}

// ---------------------------------------------------------------------------
// y = x @ W^T with bf16 WMMA, f32 accumulation, async double-buffered LDS.
//   X: [M, 1024] bf16 row-major.  W: [1024, 1024] bf16 row-major (nn.Linear).
//   Block = 256 threads (8 waves), block tile 128x64, wave tile 16x64,
//   K step 32, 4 WMMA per wave per step, pipeline unrolled by 2 so the
//   double-buffer index is a compile-time constant (no accumulator phi-copies).
//   mode 0: out bf16, head layout [B,H,N,D]
//   mode 1: out bf16, transposed head layout [B,H,D,N]   (for V)
//   mode 2: out fp32 [M, C] with bias                     (final projection)
// ---------------------------------------------------------------------------
#define A_ELEMS (128 * 40)   // stride 40 (80B): 16B-aligned rows, conflict-free
#define B_ELEMS (64 * 40)

static __device__ __forceinline__ void gemm_issue(const ushort_t* xg, const ushort_t* wg,
                                                  unsigned ab, unsigned bb) {
  async_cp16(xg,     ab);        // A tile: 2 x 16B per thread (128x32 bf16)
  async_cp16(xg + 8, ab + 16);
  async_cp16(wg,     bb);        // B tile: 1 x 16B per thread (64x32 bf16)
}

static __device__ __forceinline__ void gemm_step(const ushort_t* __restrict__ Ab,
                                                 const ushort_t* __restrict__ Bb,
                                                 int ar, int ln, int hi, v8f (&acc)[4]) {
  v8bf a0 = *(const v8bf*)&Ab[ar * 40 + hi * 16];       // K = hi*16 + 0..7
  v8bf a1 = *(const v8bf*)&Ab[ar * 40 + hi * 16 + 8];   // K = hi*16 + 8..15
  v16bf af = SHUF16(a0, a1);
#pragma unroll
  for (int tt = 0; tt < 4; ++tt) {
    const int br = tt * 16 + ln;                        // B col (N)
    v8bf b0 = *(const v8bf*)&Bb[br * 40 + hi * 16];     // K = hi*16 + 0..15
    v8bf b1 = *(const v8bf*)&Bb[br * 40 + hi * 16 + 8];
    acc[tt] = WMMA_BF16(af, SHUF16(b0, b1), acc[tt]);
  }
}

__global__ __launch_bounds__(256) void gemm_bf16_wmma(
    const ushort_t* __restrict__ X, const ushort_t* __restrict__ W,
    ushort_t* __restrict__ outB, float* __restrict__ outF,
    const float* __restrict__ bias, int mode) {
  const int K = CC;
  __shared__ ushort_t As[2][A_ELEMS];
  __shared__ ushort_t Bs[2][B_ELEMS];

  const int t    = threadIdx.x;
  const int lane = t & 31;
  const int w    = t >> 5;      // 8 waves: wave w owns rows w*16..w*16+15
  const int hi   = lane >> 4;   // half-wave select
  const int ln   = lane & 15;
  const int mBase = blockIdx.x * 128;
  const int cBase = blockIdx.y * 64;

  // async staging assignment (per K step: A tile 128x32, B tile 64x32 bf16)
  const int arow = t >> 1, acol = (t & 1) * 16;  // thread: 2x16B of A
  const int brow = t >> 2, bcol = (t & 3) * 8;   // thread: 1x16B of B
  const ushort_t* xg = X + (size_t)(mBase + arow) * K + acol;
  const ushort_t* wg = W + (size_t)(cBase + brow) * K + bcol;
  const unsigned aoff0 = lds_off(&As[0][arow * 40 + acol]);
  const unsigned boff0 = lds_off(&Bs[0][brow * 40 + bcol]);
  const unsigned aoff1 = lds_off(&As[1][arow * 40 + acol]);
  const unsigned boff1 = lds_off(&Bs[1][brow * 40 + bcol]);

  v8f acc[4] = {v8f{}, v8f{}, v8f{}, v8f{}};
  const int ar = w * 16 + ln;   // A fragment row

  // prologue: tile 0 -> buffer 0
  gemm_issue(xg, wg, aoff0, boff0);

  // pipeline: tiles 0..29 in pairs; buffer index is compile-time constant
  for (int ii = 0; ii < 15; ++ii) {
    const int kk = ii * 64;
    gemm_issue(xg + kk + 32, wg + kk + 32, aoff1, boff1);   // tile 2ii+1
    asm volatile("s_wait_asynccnt 0x3" ::: "memory");       // tile 2ii done
    __syncthreads();
    gemm_step(&As[0][0], &Bs[0][0], ar, ln, hi, acc);       // tile 2ii
    __syncthreads();
    gemm_issue(xg + kk + 64, wg + kk + 64, aoff0, boff0);   // tile 2ii+2
    asm volatile("s_wait_asynccnt 0x3" ::: "memory");       // tile 2ii+1 done
    __syncthreads();
    gemm_step(&As[1][0], &Bs[1][0], ar, ln, hi, acc);       // tile 2ii+1
    __syncthreads();
  }
  // epilogue: tiles 30 (buf0, already issued) and 31 (buf1)
  gemm_issue(xg + 31 * 32, wg + 31 * 32, aoff1, boff1);
  asm volatile("s_wait_asynccnt 0x3" ::: "memory");
  __syncthreads();
  gemm_step(&As[0][0], &Bs[0][0], ar, ln, hi, acc);         // tile 30
  asm volatile("s_wait_asynccnt 0x0" ::: "memory");
  __syncthreads();
  gemm_step(&As[1][0], &Bs[1][0], ar, ln, hi, acc);         // tile 31

  // D layout: lane holds rows (hi*8 + j), column ln of each 16x16 tile
  const int rowBase = mBase + w * 16 + hi * 8;
  if (mode == 2) {
#pragma unroll
    for (int tt = 0; tt < 4; ++tt) {
      const int c = cBase + tt * 16 + ln;
      const float bv = bias[c];
#pragma unroll
      for (int j = 0; j < 8; ++j)
        outF[(size_t)(rowBase + j) * CC + c] = acc[tt][j] + bv;
    }
  } else if (mode == 0) {
#pragma unroll
    for (int tt = 0; tt < 4; ++tt) {
      const int c = cBase + tt * 16 + ln;
      const int h = c >> 6, d = c & 63;
#pragma unroll
      for (int j = 0; j < 8; ++j) {
        const int m = rowBase + j;
        const int b = m >> 11, nn = m & (NN - 1);
        outB[(((size_t)b * HH + h) * NN + nn) * DD + d] = f2bf(acc[tt][j]);
      }
    }
  } else {  // mode 1: V transposed [B,H,D,N]
#pragma unroll
    for (int tt = 0; tt < 4; ++tt) {
      const int c = cBase + tt * 16 + ln;
      const int h = c >> 6, d = c & 63;
#pragma unroll
      for (int j = 0; j < 8; ++j) {
        const int m = rowBase + j;
        const int b = m >> 11, nn = m & (NN - 1);
        outB[(((size_t)b * HH + h) * DD + d) * NN + nn] = f2bf(acc[tt][j]);
      }
    }
  }
}

// ---------------------------------------------------------------------------
// Fused flash attention, one (b,h) x 64-query tile per block, 4 waves.
//   Q,K: [B,H,N,D] bf16.  VT: [B,H,D,N] bf16.  ctx out: [B,N,C] bf16.
//   Per key tile of 32: 4 WMMA for QK^T, online softmax (shfl row reduce),
//   P bounced through LDS (D-layout -> A-layout), 4 WMMA for P@V.
// ---------------------------------------------------------------------------
__global__ __launch_bounds__(128) void flash_attn_wmma(
    const ushort_t* __restrict__ Q, const ushort_t* __restrict__ Kb,
    const ushort_t* __restrict__ VT, ushort_t* __restrict__ ctx) {
  __shared__ ushort_t Plds[4][16 * 40];

  const int t    = threadIdx.x;
  const int lane = t & 31;
  const int w    = t >> 5;
  const int hi   = lane >> 4;
  const int ln   = lane & 15;
  const int bh   = blockIdx.y;
  const int b    = bh >> 4, h = bh & 15;
  const int qRow0 = blockIdx.x * 64 + w * 16;
  const float scale = 0.125f;  // D^-0.5, D=64

  // Q A-fragments (K=0..31 and K=32..63 of head dim)
  const ushort_t* qrow = Q + ((size_t)bh * NN + qRow0 + ln) * DD;
  v16bf qf0 = SHUF16(*(const v8bf*)(qrow + hi * 16),      *(const v8bf*)(qrow + hi * 16 + 8));
  v16bf qf1 = SHUF16(*(const v8bf*)(qrow + 32 + hi * 16), *(const v8bf*)(qrow + 32 + hi * 16 + 8));

  v8f o0 = {}, o1 = {}, o2 = {}, o3 = {};
  float mrow[8], lrow[8];
  for (int j = 0; j < 8; ++j) { mrow[j] = -1e30f; lrow[j] = 0.0f; }

  ushort_t* Pw = Plds[w];
  const ushort_t* kbh = Kb + (size_t)bh * NN * DD;
  const ushort_t* vbh = VT + (size_t)bh * DD * NN;

  for (int kt = 0; kt < NN; kt += 32) {
    // ---- scores: S[16 q rows][32 keys] as two 16x16 f32 tiles ----
    v8f s0 = {}, s1 = {};
    {
      const ushort_t* kr = kbh + (size_t)(kt + ln) * DD;      // key = kt + ln
      v16bf kb = SHUF16(*(const v8bf*)(kr + hi * 16), *(const v8bf*)(kr + hi * 16 + 8));
      s0 = WMMA_BF16(qf0, kb, s0);
      kb = SHUF16(*(const v8bf*)(kr + 32 + hi * 16), *(const v8bf*)(kr + 32 + hi * 16 + 8));
      s0 = WMMA_BF16(qf1, kb, s0);
    }
    {
      const ushort_t* kr = kbh + (size_t)(kt + 16 + ln) * DD; // key = kt + 16 + ln
      v16bf kb = SHUF16(*(const v8bf*)(kr + hi * 16), *(const v8bf*)(kr + hi * 16 + 8));
      s1 = WMMA_BF16(qf0, kb, s1);
      kb = SHUF16(*(const v8bf*)(kr + 32 + hi * 16), *(const v8bf*)(kr + 32 + hi * 16 + 8));
      s1 = WMMA_BF16(qf1, kb, s1);
    }

    // ---- online softmax; lane holds rows hi*8+j, key column ln / ln+16 ----
    for (int j = 0; j < 8; ++j) {
      float x0 = s0[j] * scale;
      float x1 = s1[j] * scale;
      float mx = fmaxf(x0, x1);
      mx = fmaxf(mx, __shfl_xor(mx, 1));
      mx = fmaxf(mx, __shfl_xor(mx, 2));
      mx = fmaxf(mx, __shfl_xor(mx, 4));
      mx = fmaxf(mx, __shfl_xor(mx, 8));   // row max across 16 key lanes
      const float mn   = fmaxf(mrow[j], mx);
      const float corr = __expf(mrow[j] - mn);
      mrow[j] = mn;
      const float p0 = __expf(x0 - mn);
      const float p1 = __expf(x1 - mn);
      float rs = p0 + p1;
      rs += __shfl_xor(rs, 1);
      rs += __shfl_xor(rs, 2);
      rs += __shfl_xor(rs, 4);
      rs += __shfl_xor(rs, 8);             // row sum
      lrow[j] = lrow[j] * corr + rs;
      o0[j] *= corr; o1[j] *= corr; o2[j] *= corr; o3[j] *= corr;
      // P in D-layout -> LDS (natural [row][key], stride 40)
      Pw[(hi * 8 + j) * 40 + ln]      = f2bf(p0);
      Pw[(hi * 8 + j) * 40 + 16 + ln] = f2bf(p1);
    }
    asm volatile("s_wait_dscnt 0x0" ::: "memory");  // wave-local LDS RAW guard

    // re-read P in A-layout: row ln, K chunks hi*16 and hi*16+8
    v16bf pf = SHUF16(*(const v8bf*)&Pw[ln * 40 + hi * 16],
                      *(const v8bf*)&Pw[ln * 40 + hi * 16 + 8]);

    // ---- O += P @ V, V B-fragments from transposed layout (contiguous in key)
    {
      const ushort_t* vr = vbh + (size_t)(0 * 16 + ln) * NN + kt;
      v16bf vb = SHUF16(*(const v8bf*)(vr + hi * 16), *(const v8bf*)(vr + hi * 16 + 8));
      o0 = WMMA_BF16(pf, vb, o0);
    }
    {
      const ushort_t* vr = vbh + (size_t)(1 * 16 + ln) * NN + kt;
      v16bf vb = SHUF16(*(const v8bf*)(vr + hi * 16), *(const v8bf*)(vr + hi * 16 + 8));
      o1 = WMMA_BF16(pf, vb, o1);
    }
    {
      const ushort_t* vr = vbh + (size_t)(2 * 16 + ln) * NN + kt;
      v16bf vb = SHUF16(*(const v8bf*)(vr + hi * 16), *(const v8bf*)(vr + hi * 16 + 8));
      o2 = WMMA_BF16(pf, vb, o2);
    }
    {
      const ushort_t* vr = vbh + (size_t)(3 * 16 + ln) * NN + kt;
      v16bf vb = SHUF16(*(const v8bf*)(vr + hi * 16), *(const v8bf*)(vr + hi * 16 + 8));
      o3 = WMMA_BF16(pf, vb, o3);
    }
  }

  // ---- epilogue: normalize, store ctx [B,N,C] bf16 (c = h*64 + d) ----
  const size_t rowB = (size_t)b * NN + qRow0 + hi * 8;
  for (int j = 0; j < 8; ++j) {
    const float inv = 1.0f / lrow[j];
    const size_t base = (rowB + j) * CC + (size_t)h * 64 + ln;
    ctx[base +  0] = f2bf(o0[j] * inv);
    ctx[base + 16] = f2bf(o1[j] * inv);
    ctx[base + 32] = f2bf(o2[j] * inv);
    ctx[base + 48] = f2bf(o3[j] * inv);
  }
}

// ---------------------------------------------------------------------------
extern "C" void kernel_launch(void* const* d_in, const int* in_sizes, int n_in,
                              void* d_out, int out_size, void* d_ws, size_t ws_size,
                              hipStream_t stream) {
  (void)in_sizes; (void)n_in; (void)out_size; (void)ws_size;
  const float* query  = (const float*)d_in[0];
  const float* target = (const float*)d_in[1];
  const float* Wq = (const float*)d_in[2];
  const float* Wk = (const float*)d_in[3];
  const float* Wv = (const float*)d_in[4];
  const float* Wp = (const float*)d_in[5];
  const float* bp = (const float*)d_in[6];
  float* out = (float*)d_out;

  const size_t ACT = (size_t)BB * NN * CC;  // 4M elements
  const size_t WEL = (size_t)CC * CC;       // 1M elements
  ushort_t* ws    = (ushort_t*)d_ws;        // total use: 28M bf16 = 56 MB
  ushort_t* qry16 = ws;
  ushort_t* tgt16 = qry16 + ACT;
  ushort_t* wq16  = tgt16 + ACT;
  ushort_t* wk16  = wq16 + WEL;
  ushort_t* wv16  = wk16 + WEL;
  ushort_t* wp16  = wv16 + WEL;
  ushort_t* qh    = wp16 + WEL;   // [B,H,N,D]
  ushort_t* kh    = qh + ACT;     // [B,H,N,D]
  ushort_t* vT    = kh + ACT;     // [B,H,D,N]
  ushort_t* ctx   = vT + ACT;     // [B,N,C]

  cvt_f32_bf16<<<(ACT + 255) / 256, 256, 0, stream>>>(query,  qry16, (int)ACT);
  cvt_f32_bf16<<<(ACT + 255) / 256, 256, 0, stream>>>(target, tgt16, (int)ACT);
  cvt_f32_bf16<<<(WEL + 255) / 256, 256, 0, stream>>>(Wq, wq16, (int)WEL);
  cvt_f32_bf16<<<(WEL + 255) / 256, 256, 0, stream>>>(Wk, wk16, (int)WEL);
  cvt_f32_bf16<<<(WEL + 255) / 256, 256, 0, stream>>>(Wv, wv16, (int)WEL);
  cvt_f32_bf16<<<(WEL + 255) / 256, 256, 0, stream>>>(Wp, wp16, (int)WEL);

  dim3 gg(BB * NN / 128, CC / 64);  // (32, 16)
  gemm_bf16_wmma<<<gg, 256, 0, stream>>>(qry16, wq16, qh, nullptr, nullptr, 0);
  gemm_bf16_wmma<<<gg, 256, 0, stream>>>(tgt16, wk16, kh, nullptr, nullptr, 0);
  gemm_bf16_wmma<<<gg, 256, 0, stream>>>(tgt16, wv16, vT, nullptr, nullptr, 1);

  flash_attn_wmma<<<dim3(NN / 64, BB * HH), 128, 0, stream>>>(qh, kh, vT, ctx);

  gemm_bf16_wmma<<<gg, 256, 0, stream>>>(ctx, wp16, nullptr, out, bp, 2);
}
